// SettleCharLM_24919400251463
// MI455X (gfx1250) — compile-verified, matrix operand order checked by reading, not run on hardware
//
#include <hip/hip_runtime.h>
#include <hip/hip_bf16.h>

#define D        256
#define DH       1024
#define N_LAYERS 4
#define K_SETTLE 2
#define VOCAB    256
#define BB       128
#define TT       1024
#define EPSV     1e-8f
#define ALPHAV   0.2f

typedef __attribute__((ext_vector_type(16))) __bf16 v16bf;
typedef __attribute__((ext_vector_type(8)))  float  v8f;

#define V8F_ZERO {0.f, 0.f, 0.f, 0.f, 0.f, 0.f, 0.f, 0.f}

union FragU { v16bf v; uint4 q[2]; };

// Opaque zero offset: volatile empty asm blocks LICM from hoisting the weight
// fragment loads out of the persistent time loop (register spills), while the
// base pointer chain stays intact so address-space inference still yields
// global_load_b128 (not flat_load, which would pollute DScnt).
__device__ __forceinline__ size_t opaque_zero() {
  unsigned long long z = 0;
  asm volatile("" : "+s"(z));
  return (size_t)z;
}

// ---- WMMA fragment loaders (16-bit A/B layout per CDNA5 ISA 7.12.2) ----
// lane<16 : elements 0..7 = K[kk..kk+7],   8..15 = K[kk+16..kk+23]
// lane>=16: elements 0..7 = K[kk+8..kk+15],8..15 = K[kk+24..kk+31]
__device__ __forceinline__ v16bf ldA(const __bf16* base, int ld, int kk, int lane) {
  int m  = lane & 15;
  int ko = kk + ((lane >> 4) << 3);
  FragU f;
  f.q[0] = *(const uint4*)(base + m * ld + ko);
  f.q[1] = *(const uint4*)(base + m * ld + ko + 16);
  return f.v;
}
// B fragment: per-lane column n of B == row n of row-major weight W[N][K]
__device__ __forceinline__ v16bf ldB(const __bf16* w, int ldk, int n0, int kk, int lane) {
  int n  = n0 + (lane & 15);
  int ko = kk + ((lane >> 4) << 3);
  FragU f;
  f.q[0] = *(const uint4*)(w + (size_t)n * ldk + ko);
  f.q[1] = *(const uint4*)(w + (size_t)n * ldk + ko + 16);
  return f.v;
}

__device__ __forceinline__ v8f wmma_bf16(v16bf a, v16bf b, v8f c) {
  return __builtin_amdgcn_wmma_f32_16x16x32_bf16(false, a, false, b, (short)0, c, false, false);
}

// Preload all 8 A fragments of a K=256 GEMM (64 VGPRs, reused across tiles)
__device__ __forceinline__ void ldA_all256(const __bf16* A, int lda, int lane, v16bf* fr) {
#pragma unroll
  for (int i = 0; i < 8; ++i) fr[i] = ldA(A, lda, i * 32, lane);
}

// K=256 GEMM tile with preloaded A; B stream double-buffered
__device__ __forceinline__ v8f gemm256_pa(const v16bf* aw, const __bf16* W,
                                          int ldk, int n0, int lane, v8f acc) {
  v16bf b0 = ldB(W, ldk, n0, 0, lane);
#pragma unroll
  for (int i = 0; i < 7; ++i) {
    v16bf b1 = ldB(W, ldk, n0, (i + 1) * 32, lane);
    acc = wmma_bf16(aw[i], b0, acc);
    b0 = b1;
  }
  return wmma_bf16(aw[7], b0, acc);
}

__device__ __forceinline__ float wave_sum(float v) {
#pragma unroll
  for (int off = 16; off > 0; off >>= 1) v += __shfl_xor(v, off, 32);
  return v;
}
__device__ __forceinline__ float sigm(float x) { return 1.f / (1.f + __expf(-x)); }

// ---------------- weight f32 -> bf16 conversion ----------------
__global__ void cvt_f32_bf16(const float* __restrict__ src, __bf16* __restrict__ dst, int n) {
  int i = blockIdx.x * blockDim.x + threadIdx.x;
  if (i < n) dst[i] = (__bf16)src[i];
}

// ---------------- persistent recurrent kernel: 8 WGs x 16 batch rows ----------------
__global__ void __launch_bounds__(512, 1)
settle_lm_kernel(const int* __restrict__ x, const float* __restrict__ tok_emb,
                 const float* __restrict__ state_norm_scale,
                 const float* __restrict__ state_gate_b,
                 const float* __restrict__ settle_gate_b,
                 const float* __restrict__ norm_scales,
                 const float* __restrict__ fc1_b, const float* __restrict__ fc2_b,
                 const float* __restrict__ out_norm_scale,
                 const __bf16* __restrict__ w_sg, const __bf16* __restrict__ w_tg,
                 const __bf16* __restrict__ w_lm, const __bf16* __restrict__ w_fc1,
                 const __bf16* __restrict__ w_fc2,
                 float* __restrict__ state_ws, float* __restrict__ out) {
  __shared__ float  sh_hw[16][D];          // working h (f32)          16 KB
  __shared__ float  sh_prev[16][D];        // settle prev (f32)        16 KB
  __shared__ __bf16 sh_a[16][D + 8];       // bf16 A operand           8.25 KB
  __shared__ __bf16 sh_mid[16][512 + 8];   // bf16 gelu(mid) half      16.25 KB
  __shared__ int    sh_tok[16];

  const int tid   = threadIdx.x;
  const int wv    = tid >> 5;              // 16 waves, wave wv owns row wv / N-tile wv
  const int lane  = tid & 31;
  const int bBase = blockIdx.x * 16;       // 8 workgroups
  const int nlo   = lane & 15;
  const int mhi   = (lane >> 4) << 3;

  for (int t = 0; t < TT; ++t) {
    // per-timestep opaque offsets: keep weight loads inside the loop
    const __bf16* wsg = w_sg + opaque_zero();
    const __bf16* wtg = w_tg + opaque_zero();
    const __bf16* wlm = w_lm + opaque_zero();

    // ---- phase 1: token ids, state rmsnorm -> sh_a (bf16), prefetch next emb ----
    if (tid < 16) sh_tok[tid] = x[(size_t)(bBase + tid) * TT + t];
    {
      const float* srow = state_ws + (size_t)(bBase + wv) * D;
      const int k0 = lane * 8;
      float v[8], ss = 0.f;
#pragma unroll
      for (int i = 0; i < 8; ++i) { v[i] = srow[k0 + i]; ss += v[i] * v[i]; }
      ss = wave_sum(ss);
      float rms = rsqrtf(ss * (1.f / D) + EPSV);
#pragma unroll
      for (int i = 0; i < 8; ++i)
        sh_a[wv][k0 + i] = (__bf16)(v[i] * rms * state_norm_scale[k0 + i]);
    }
    if (tid < 16 && t + 1 < TT) {
      int tk = x[(size_t)(bBase + tid) * TT + t + 1];
      const char* p = (const char*)(tok_emb + (size_t)tk * D);
      __builtin_prefetch(p);       __builtin_prefetch(p + 256);
      __builtin_prefetch(p + 512); __builtin_prefetch(p + 768);
    }
    __syncthreads();

    // ---- phase 2: state gate GEMM; h = emb + sigmoid(.)*s_in -> hw, prev ----
    {
      v16bf aw[8];
      ldA_all256(&sh_a[0][0], D + 8, lane, aw);
      v8f acc = V8F_ZERO;
      acc = gemm256_pa(aw, wsg, D, wv * 16, lane, acc);
      float bias = state_gate_b[wv * 16 + nlo];
#pragma unroll
      for (int r = 0; r < 8; ++r) {
        int m = r + mhi, n = (wv << 4) + nlo;
        float g  = sigm(acc[r] + bias);
        float si = (float)sh_a[m][n];
        float h  = tok_emb[(size_t)sh_tok[m] * D + n] + g * si;
        sh_hw[m][n] = h; sh_prev[m][n] = h;
      }
    }
    __syncthreads();

    // ---- settling iterations ----
    for (int ks = 0; ks < K_SETTLE; ++ks) {
      // per-iteration opaque offsets (fc weights invariant across ks)
      const __bf16* wf1base = w_fc1 + opaque_zero();
      const __bf16* wf2base = w_fc2 + opaque_zero();
      for (int l = 0; l < N_LAYERS; ++l) {
        // a) rmsnorm(hw) * scale -> sh_a
        {
          const int k0 = lane * 8;
          float v[8], ss = 0.f;
#pragma unroll
          for (int i = 0; i < 8; ++i) { v[i] = sh_hw[wv][k0 + i]; ss += v[i] * v[i]; }
          ss = wave_sum(ss);
          float rms = rsqrtf(ss * (1.f / D) + EPSV);
          const float* sc = norm_scales + (size_t)l * D;
#pragma unroll
          for (int i = 0; i < 8; ++i)
            sh_a[wv][k0 + i] = (__bf16)(v[i] * rms * sc[k0 + i]);
        }
        __syncthreads();

        v8f acc2 = V8F_ZERO;
        const __bf16* Wf1 = wf1base + (size_t)l * DH * D;
        const __bf16* Wf2 = wf2base + (size_t)l * D * DH;
        for (int hh = 0; hh < 2; ++hh) {
          // b) fc1 half: wave owns tiles nt0,nt1 sharing preloaded A fragments.
          //    Per K-step: 2 pipelined ldB + 2 WMMAs back-to-back.
          {
            int nt0 = hh * 32 + wv;
            int nt1 = hh * 32 + 16 + wv;
            v16bf aw[8];
            ldA_all256(&sh_a[0][0], D + 8, lane, aw);
            v8f a1 = V8F_ZERO, a2 = V8F_ZERO;
            v16bf bf0 = ldB(Wf1, D, nt0 * 16, 0, lane);
            v16bf bf1 = ldB(Wf1, D, nt1 * 16, 0, lane);
#pragma unroll
            for (int i = 0; i < 7; ++i) {
              v16bf bf0n = ldB(Wf1, D, nt0 * 16, (i + 1) * 32, lane);
              v16bf bf1n = ldB(Wf1, D, nt1 * 16, (i + 1) * 32, lane);
              a1 = wmma_bf16(aw[i], bf0, a1);
              a2 = wmma_bf16(aw[i], bf1, a2);
              bf0 = bf0n; bf1 = bf1n;
            }
            a1 = wmma_bf16(aw[7], bf0, a1);
            a2 = wmma_bf16(aw[7], bf1, a2);
            float b1 = fc1_b[(size_t)l * DH + nt0 * 16 + nlo];
            float b2 = fc1_b[(size_t)l * DH + nt1 * 16 + nlo];
#pragma unroll
            for (int r = 0; r < 8; ++r) {
              int m  = r + mhi;
              int c0 = wv * 16 + nlo;           // (nt0 - hh*32)*16
              int c1 = (16 + wv) * 16 + nlo;    // (nt1 - hh*32)*16
              float x0 = a1[r] + b1;
              float x1 = a2[r] + b2;
              sh_mid[m][c0] = (__bf16)(0.5f * x0 * (1.f + erff(x0 * 0.70710678118f)));
              sh_mid[m][c1] = (__bf16)(0.5f * x1 * (1.f + erff(x1 * 0.70710678118f)));
            }
          }
          __syncthreads();
          // c) fc2 partial: wave wv owns output tile wv; pipelined K=512 half
          {
            v16bf af = ldA(&sh_mid[0][0], 512 + 8, 0, lane);
            v16bf bf = ldB(Wf2, DH, wv * 16, hh * 512, lane);
#pragma unroll 4
            for (int kk = 0; kk < 480; kk += 32) {
              v16bf afn = ldA(&sh_mid[0][0], 512 + 8, kk + 32, lane);
              v16bf bfn = ldB(Wf2, DH, wv * 16, hh * 512 + kk + 32, lane);
              acc2 = wmma_bf16(af, bf, acc2);
              af = afn; bf = bfn;
            }
            acc2 = wmma_bf16(af, bf, acc2);
          }
          __syncthreads();
        }
        // d) residual: hw += fc2_out + bias (per-wave owned positions)
        float b2 = fc2_b[(size_t)l * D + wv * 16 + nlo];
#pragma unroll
        for (int r = 0; r < 8; ++r) {
          int m = r + mhi, n = (wv << 4) + nlo;
          sh_hw[m][n] += acc2[r] + b2;
        }
        __syncthreads();
      }
      // settle gate: sh_a = bf16(hc); h = prev + sigmoid(hc@Wt.T+b)*(hc-prev)
      {
        const int k0 = lane * 8;
#pragma unroll
        for (int i = 0; i < 8; ++i) sh_a[wv][k0 + i] = (__bf16)sh_hw[wv][k0 + i];
      }
      __syncthreads();
      {
        v16bf aw[8];
        ldA_all256(&sh_a[0][0], D + 8, lane, aw);
        v8f acc = V8F_ZERO;
        acc = gemm256_pa(aw, wtg, D, wv * 16, lane, acc);
        float bs = settle_gate_b[wv * 16 + nlo];
#pragma unroll
        for (int r = 0; r < 8; ++r) {
          int m = r + mhi, n = (wv << 4) + nlo;
          float hc = sh_hw[m][n], pv = sh_prev[m][n];
          float hn = pv + sigm(acc[r] + bs) * (hc - pv);
          sh_hw[m][n] = hn; sh_prev[m][n] = hn;
        }
      }
      __syncthreads();
    }

    // ---- state EMA update + output rmsnorm -> sh_a ----
    {
      float* srow = state_ws + (size_t)(bBase + wv) * D;
      const int k0 = lane * 8;
      float v[8], ss = 0.f;
#pragma unroll
      for (int i = 0; i < 8; ++i) {
        float h = sh_hw[wv][k0 + i];
        v[i] = h; ss += h * h;
        srow[k0 + i] = (1.f - ALPHAV) * srow[k0 + i] + ALPHAV * h;
      }
      ss = wave_sum(ss);
      float rms = rsqrtf(ss * (1.f / D) + EPSV);
#pragma unroll
      for (int i = 0; i < 8; ++i)
        sh_a[wv][k0 + i] = (__bf16)(v[i] * rms * out_norm_scale[k0 + i]);
    }
    __syncthreads();

    // ---- lm head: wave wv -> vocab tile wv; nontemporal f32 logit stream ----
    {
      v16bf aw[8];
      ldA_all256(&sh_a[0][0], D + 8, lane, aw);
      v8f acc = V8F_ZERO;
      acc = gemm256_pa(aw, wlm, D, wv * 16, lane, acc);
#pragma unroll
      for (int r = 0; r < 8; ++r) {
        int m = r + mhi, n = (wv << 4) + nlo;
        // 128 MB write-once stream: NT stores keep weights resident in L2
        __builtin_nontemporal_store(acc[r],
            &out[((size_t)(bBase + m) * TT + t) * VOCAB + n]);
      }
    }
    __syncthreads();  // protect sh_a before next timestep's phase 1
  }
}

extern "C" void kernel_launch(void* const* d_in, const int* in_sizes, int n_in,
                              void* d_out, int out_size, void* d_ws, size_t ws_size,
                              hipStream_t stream) {
  (void)in_sizes; (void)n_in; (void)out_size; (void)ws_size;
  const int*   x          = (const int*)d_in[0];
  const float* state0     = (const float*)d_in[1];
  const float* tok_emb    = (const float*)d_in[2];
  const float* st_scale   = (const float*)d_in[3];
  const float* sg_w       = (const float*)d_in[4];
  const float* sg_b       = (const float*)d_in[5];
  const float* tg_w       = (const float*)d_in[6];
  const float* tg_b       = (const float*)d_in[7];
  const float* nscales    = (const float*)d_in[8];
  const float* fc1_w      = (const float*)d_in[9];
  const float* fc1_b      = (const float*)d_in[10];
  const float* fc2_w      = (const float*)d_in[11];
  const float* fc2_b      = (const float*)d_in[12];
  const float* out_scale  = (const float*)d_in[13];
  const float* lm_w       = (const float*)d_in[14];
  float* out = (float*)d_out;

  // workspace layout
  float*  state_ws = (float*)d_ws;                                   // 128*256 f32
  __bf16* b_sg  = (__bf16*)((char*)d_ws + (size_t)BB * D * 4);
  __bf16* b_tg  = b_sg  + (size_t)D * D;
  __bf16* b_lm  = b_tg  + (size_t)D * D;
  __bf16* b_fc1 = b_lm  + (size_t)VOCAB * D;
  __bf16* b_fc2 = b_fc1 + (size_t)N_LAYERS * DH * D;

  // reset carried state (deterministic per call)
  hipMemcpyAsync(state_ws, state0, (size_t)BB * D * sizeof(float),
                 hipMemcpyDeviceToDevice, stream);

  // one-shot weight conversion to bf16 (L2-resident thereafter)
  auto cvt = [&](const float* s, __bf16* d, int n) {
    cvt_f32_bf16<<<(n + 255) / 256, 256, 0, stream>>>(s, d, n);
  };
  cvt(sg_w,  b_sg,  D * D);
  cvt(tg_w,  b_tg,  D * D);
  cvt(lm_w,  b_lm,  VOCAB * D);
  cvt(fc1_w, b_fc1, N_LAYERS * DH * D);
  cvt(fc2_w, b_fc2, N_LAYERS * D * DH);

  // persistent recurrent kernel: 8 workgroups (one per 16-row batch tile), 16 waves each
  settle_lm_kernel<<<BB / 16, 512, 0, stream>>>(
      x, tok_emb, st_scale, sg_b, tg_b, nscales, fc1_b, fc2_b, out_scale,
      b_sg, b_tg, b_lm, b_fc1, b_fc2, state_ws, out);
}
